// HyperbolicGraphConvolution_29532195127738
// MI455X (gfx1250) — matrix operand order repaired and checked by reading, not compiled, not run on hardware
//
#include <hip/hip_runtime.h>
#include <hip/hip_bf16.h>
#include <math.h>

// CDNA5 / gfx1250: wave32, WMMA 16x16x32 bf16 (fp32 accumulate).
typedef __attribute__((ext_vector_type(16))) __bf16 v16bf;
typedef __attribute__((ext_vector_type(8)))  float  v8f;

#define N_ROWS 16384
#define DIM    128
#define NK     (N_ROWS / 32)      // 512 K-steps of 32
#define MINN   1e-15f
#define MAXN   (1.0f - 4e-3f)     // (1 - BALL_EPS)/sqrt(c), c = 1
#define ATCLIP (1.0f - 1e-7f)

__device__ __forceinline__ float artanh_clip(float x) {
    x = fminf(x, ATCLIP);                      // args here are always >= 0
    return 0.5f * logf((1.0f + x) / (1.0f - x));
}

// block = 128 threads = 4 waves; full-block sum, result broadcast to all threads
__device__ __forceinline__ float blockReduceSum(float v, float* buf) {
    #pragma unroll
    for (int m = 16; m >= 1; m >>= 1) v += __shfl_xor(v, m, 32);
    const int w = threadIdx.x >> 5;
    if ((threadIdx.x & 31) == 0) buf[w] = v;
    __syncthreads();
    v = buf[0] + buf[1] + buf[2] + buf[3];
    __syncthreads();
    return v;
}

__device__ __forceinline__ unsigned pk2bf(float a, float b) {  // round-to-nearest-even
    unsigned ua = __float_as_uint(a), ub = __float_as_uint(b);
    ua += 0x7FFFu + ((ua >> 16) & 1u);
    ub += 0x7FFFu + ((ub >> 16) & 1u);
    return (ua >> 16) | (ub & 0xFFFF0000u);
}

// ---------------------------------------------------------------------------
// Kernel 1: HypLinear + logmap0.  Output xt written as bf16 in the WMMA
// B-fragment layout: fragment f = (row/32)*8 + (col/16), 1024 B per fragment,
// lane = (row%32 / 16)*16 + col%16, dword = (row%16)/2, half = row%2.
// ---------------------------------------------------------------------------
__global__ __launch_bounds__(128) void hyp_linear_logmap(
    const float* __restrict__ x, const float* __restrict__ W,
    const float* __restrict__ bias, unsigned short* __restrict__ xtB)
{
    __shared__ float Ws[DIM * (DIM + 1)];   // stride-129: (t+k)%64 banks, conflict-free
    __shared__ float xrow[DIM];
    __shared__ float hb[DIM];
    __shared__ float red[4];

    const int t = threadIdx.x;

    for (int i = t; i < DIM * DIM; i += DIM)
        Ws[(i >> 7) * (DIM + 1) + (i & 127)] = W[i];

    // hyp_bias = proj(expmap0(bias))  (uniform across the block)
    float b   = bias[t];
    float bn2 = blockReduceSum(b * b, red);
    float bn  = fmaxf(sqrtf(bn2), MINN);
    float tb  = tanhf(bn);
    float bf  = tb / bn;
    float hbn = tb;
    if (tb > MAXN) { bf = MAXN / bn; hbn = MAXN; }
    hb[t] = bf * b;
    const float y2 = hbn * hbn;
    __syncthreads();
    const float hbv = hb[t];

    const int r0 = blockIdx.x * 32;
    for (int rr = 0; rr < 32; ++rr) {
        const int r = r0 + rr;
        xrow[t] = x[(size_t)r * DIM + t];
        __syncthreads();
        const float xv = xrow[t];
        const float x2 = blockReduceSum(xv * xv, red);

        float mx = 0.f;
        #pragma unroll 8
        for (int k = 0; k < DIM; ++k) mx += Ws[t * (DIM + 1) + k] * xrow[k];
        const float mxn2 = blockReduceSum(mx * mx, red);

        // mobius_matvec + proj
        const float xn  = fmaxf(sqrtf(x2), MINN);
        const float mxn = fmaxf(sqrtf(mxn2), MINN);
        const float th  = tanhf((mxn / xn) * artanh_clip(xn));
        float rf   = th / mxn;
        float resn = fabsf(th);
        if (resn > MAXN) { rf *= MAXN / resn; resn = MAXN; }
        const float res = rf * mx;              // mx == 0 -> res == 0 (zero-row case)
        const float x2r = resn * resn;

        // mobius_add(res, hyp_bias) + proj
        const float xy  = blockReduceSum(res * hbv, red);
        const float cc  = 1.f + 2.f * xy + y2;
        const float den = fmaxf(1.f + 2.f * xy + x2r * y2, MINN);
        float h = (cc * res + (1.f - x2r) * hbv) / den;
        const float hn2 = blockReduceSum(h * h, red);
        float hn = fmaxf(sqrtf(hn2), MINN);
        if (hn > MAXN) { h *= MAXN / hn; hn = MAXN; }

        // logmap0
        const float xt = (artanh_clip(hn) / hn) * h;

        // scatter-store bf16 in WMMA B-fragment layout (frag stride = 1 KB)
        const int kk = r & 31, kt = r >> 5;
        const int nt = t >> 4, nn = t & 15;
        const int lane = ((kk >> 4) << 4) | nn;
        const int vr = (kk & 15) >> 1, half = kk & 1;
        const size_t off = (size_t)(kt * 8 + nt) * 1024 + (size_t)lane * 32 + vr * 4 + half * 2;
        unsigned ua = __float_as_uint(xt);
        ua += 0x7FFFu + ((ua >> 16) & 1u);
        *(unsigned short*)((char*)xtB + off) = (unsigned short)(ua >> 16);
    }
}

// ---------------------------------------------------------------------------
// Kernel 2: support_t = adj @ xt via v_wmma_f32_16x16x32_bf16, fused with
// proj(expmap0(.)), relu(logmap0(.)), proj(expmap0(.)) epilogue.
// One wave owns 16 output rows x 128 cols (8 accumulators of v8f).
// Double-buffered register pipeline: while buffer p feeds 8 WMMAs, the 20
// loads of buffer p^1 (next K-step) are in flight.  sched_barrier(0) pins
// the load group ahead of the compute group so the scheduler cannot sink
// the loads back to their uses (which would re-serialize on s_wait_loadcnt).
// ---------------------------------------------------------------------------
__global__ __launch_bounds__(128) void hyp_agg_act(
    const float* __restrict__ adj, const unsigned short* __restrict__ xtB,
    float* __restrict__ out)
{
    const int lane = threadIdx.x & 31;
    const int wave = (blockIdx.x * blockDim.x + threadIdx.x) >> 5;   // 0..1023
    const int m0   = wave * 16;
    const int mrow = m0 + (lane & 15);
    const int clo  = (lane < 16) ? 0 : 16;                // A-layout K split per lane half
    const float4* __restrict__ A4 = (const float4*)(adj + (size_t)mrow * N_ROWS + clo);
    const uint4*  __restrict__ B4 = (const uint4*)xtB;    // frag = 64 uint4 = 1 KB

    v8f acc[8] = {};

    float4 a[2][4];     // raw fp32 A fragments, double-buffered
    uint4  bq[2][16];   // bf16 B fragments, double-buffered

    auto loadA = [&](int p, int kt) {
        a[p][0] = A4[kt * 8 + 0];
        a[p][1] = A4[kt * 8 + 1];
        a[p][2] = A4[kt * 8 + 2];
        a[p][3] = A4[kt * 8 + 3];
    };
    auto loadB = [&](int p, int kt) {
        const uint4* __restrict__ bp = B4 + (size_t)kt * 512 + lane * 2;
        #pragma unroll
        for (int nt = 0; nt < 8; ++nt) {
            bq[p][2 * nt]     = bp[nt * 64];
            bq[p][2 * nt + 1] = bp[nt * 64 + 1];
        }
    };
    auto compute = [&](int p) {
        union { unsigned u[8]; v16bf v; } Af;
        Af.u[0] = pk2bf(a[p][0].x, a[p][0].y); Af.u[1] = pk2bf(a[p][0].z, a[p][0].w);
        Af.u[2] = pk2bf(a[p][1].x, a[p][1].y); Af.u[3] = pk2bf(a[p][1].z, a[p][1].w);
        Af.u[4] = pk2bf(a[p][2].x, a[p][2].y); Af.u[5] = pk2bf(a[p][2].z, a[p][2].w);
        Af.u[6] = pk2bf(a[p][3].x, a[p][3].y); Af.u[7] = pk2bf(a[p][3].z, a[p][3].w);
        #pragma unroll
        for (int nt = 0; nt < 8; ++nt) {
            union { uint4 q[2]; v16bf v; } Bf;
            Bf.q[0] = bq[p][2 * nt];
            Bf.q[1] = bq[p][2 * nt + 1];
            acc[nt] = __builtin_amdgcn_wmma_f32_16x16x32_bf16(
                false, Af.v, false, Bf.v, (short)0, acc[nt], false, false);
        }
    };

    loadA(0, 0);
    loadB(0, 0);

    for (int kt = 0; kt < NK; kt += 2) {
        const int kt1 = (kt + 1 < NK) ? (kt + 1) : (NK - 1);
        loadA(1, kt1);
        loadB(1, kt1);
        {   // stream prefetch ~1 KB ahead on the adj row
            const int ktp = (kt + 8 < NK) ? (kt + 8) : (NK - 1);
            __builtin_prefetch((const void*)(A4 + ktp * 8), 0, 1);
        }
        __builtin_amdgcn_sched_barrier(0);
        compute(0);                               // kt
        __builtin_amdgcn_sched_barrier(0);

        const int kt2 = (kt + 2 < NK) ? (kt + 2) : (NK - 1);
        loadA(0, kt2);
        loadB(0, kt2);
        {
            const int ktp = (kt + 9 < NK) ? (kt + 9) : (NK - 1);
            __builtin_prefetch((const void*)(A4 + ktp * 8), 0, 1);
        }
        __builtin_amdgcn_sched_barrier(0);
        compute(1);                               // kt + 1
        __builtin_amdgcn_sched_barrier(0);
    }

    // Epilogue. D layout: VGPR j -> row m0 + (lane/16)*8 + j, col = nt*16 + lane%16.
    // Row-norm reductions stay inside each 16-lane half (xor masks 1,2,4,8).
    #pragma unroll
    for (int j = 0; j < 8; ++j) {
        float s2 = 0.f;
        #pragma unroll
        for (int nt = 0; nt < 8; ++nt) { const float d = acc[nt][j]; s2 += d * d; }
        #pragma unroll
        for (int m = 8; m >= 1; m >>= 1) s2 += __shfl_xor(s2, m, 32);

        // h = proj(expmap0(s))
        const float sn = fmaxf(sqrtf(s2), MINN);
        const float t1 = tanhf(sn);
        float f1 = t1 / sn;
        float hn = t1;
        if (t1 > MAXN) { f1 = MAXN / sn; hn = MAXN; }
        hn = fmaxf(hn, MINN);
        // t = relu(logmap0(h)) = (artanh(hn)/hn) * relu(h)
        const float f2 = artanh_clip(hn) / hn;
        float tv[8]; float t2 = 0.f;
        #pragma unroll
        for (int nt = 0; nt < 8; ++nt) {
            const float hc = f1 * acc[nt][j];
            const float tc = f2 * fmaxf(hc, 0.f);
            tv[nt] = tc; t2 += tc * tc;
        }
        #pragma unroll
        for (int m = 8; m >= 1; m >>= 1) t2 += __shfl_xor(t2, m, 32);

        // out = proj(expmap0(t))
        const float tn = fmaxf(sqrtf(t2), MINN);
        const float t3 = tanhf(tn);
        float f3 = t3 / tn;
        if (t3 > MAXN) f3 = MAXN / tn;

        const int row = m0 + ((lane >> 4) << 3) + j;
        const int col = lane & 15;
        #pragma unroll
        for (int nt = 0; nt < 8; ++nt)
            out[(size_t)row * DIM + nt * 16 + col] = f3 * tv[nt];
    }
}

extern "C" void kernel_launch(void* const* d_in, const int* in_sizes, int n_in,
                              void* d_out, int out_size, void* d_ws, size_t ws_size,
                              hipStream_t stream) {
    const float* x    = (const float*)d_in[0];   // [16384,128]
    const float* adj  = (const float*)d_in[1];   // [16384,16384]
    const float* W    = (const float*)d_in[2];   // [128,128]
    const float* bias = (const float*)d_in[3];   // [128]
    float* out = (float*)d_out;                  // [16384,128] fp32
    unsigned short* xtB = (unsigned short*)d_ws; // 16384*128 bf16 = 4 MB, frag layout

    hipLaunchKernelGGL(hyp_linear_logmap, dim3(N_ROWS / 32), dim3(DIM), 0, stream,
                       x, W, bias, xtB);
    hipLaunchKernelGGL(hyp_agg_act, dim3(N_ROWS / 64), dim3(128), 0, stream,
                       adj, xtB, out);
}
